// CrossAttnBlock_18305150616122
// MI455X (gfx1250) — matrix operand order repaired and checked by reading, not compiled, not run on hardware
//
#include <hip/hip_runtime.h>

// ---------------------------------------------------------------------------
// CDNA5 (gfx1250) fused cross-attention block.
//
// Roofline: mandatory HBM traffic = x (134MB f32 in) + out (134MB f32 out)
// ~= 268MB -> ~11.5us @ 23.3TB/s. One fused kernel per 128-row block:
//   TDM:    per-wave tensor_load_to_lds of K/V tiles (overlaps phase 1)
//   Phase1: Q = x @ Wq, A-fragments register-resident, barrier-free
//   Phase3: per-head attention (wave == head), O overwrites Q in LDS
//   Phase4: out = O @ Wo + bo
// All matrix math: v_wmma_f32_16x16x32_bf16 (f32 accumulate).
// ---------------------------------------------------------------------------

typedef __bf16 bf16_t;
typedef __attribute__((ext_vector_type(16))) __bf16 v16bf;
typedef __attribute__((ext_vector_type(8)))  float  v8f;
typedef __attribute__((ext_vector_type(4)))  unsigned int u32x4;
typedef __attribute__((ext_vector_type(8)))  int i32x8;
typedef __attribute__((ext_vector_type(4)))  int i32x4;

#define FEAT    512
#define COND_D  768
#define NHEADS  8
#define DHEAD   64
#define BATCH   16
#define SEQ_N   4096
#define SEQ_M   77
#define MPAD    96          // padded M in the global k/v workspace
#define ROWS_WG 128         // rows of x per workgroup

// ---- LDS layout (dynamic shared memory; all pitches keep b128 alignment) --
#define QO_PITCH 520                           // 512 + 8 (conflict-free: 1040B)
#define QO_BYTES (ROWS_WG * QO_PITCH * 2)      // 133120
#define KLDS_ROWS 77
#define KLDS_HEAD (KLDS_ROWS * DHEAD * 2)      // 9856 B per head
#define K_OFF     QO_BYTES
#define K_BYTES   (NHEADS * KLDS_HEAD)         // 78848
#define VLDS_PITCH 80                          // m in [0,80)
#define VLDS_HEAD (DHEAD * VLDS_PITCH * 2)     // 10240 B per head
#define V_OFF     (K_OFF + K_BYTES)
#define V_BYTES   (NHEADS * VLDS_HEAD)         // 81920
#define P_PITCH   96
#define P_OFF     (V_OFF + V_BYTES)
#define P_BYTES   (NHEADS * 16 * P_PITCH * 2)  // 24576
#define SMEM_BYTES (P_OFF + P_BYTES)           // 318464  (< 320KB WGP LDS)

// ---------------- bf16 helpers (round-to-nearest-even) ---------------------
__device__ __forceinline__ unsigned short bf16_bits(float f) {
  unsigned u = __builtin_bit_cast(unsigned, f);
  u = (u + 0x7FFFu + ((u >> 16) & 1u)) >> 16;
  return (unsigned short)u;
}
__device__ __forceinline__ bf16_t f2bf(float f) {
  unsigned short s = bf16_bits(f);
  return __builtin_bit_cast(bf16_t, s);
}
__device__ __forceinline__ unsigned pack2(float a, float b) {
  return (unsigned)bf16_bits(a) | ((unsigned)bf16_bits(b) << 16);
}

union FragBF {
  uint4 q[2];
  v16bf v;
};

// A-fragment (16x32 bf16, M x K). ISA 7.12.2: lanes 0-15 row m hold
// K {0..7, 16..23}; lanes 16-31 hold K {8..15, 24..31}.
__device__ __forceinline__ v16bf ld_afrag(const bf16_t* base, int pitch,
                                          int row0, int col0, int lane) {
  const int m = lane & 15, hh = lane >> 4;
  const bf16_t* p = base + (size_t)(row0 + m) * pitch + col0;
  FragBF f;
  f.q[0] = *(const uint4*)(p + 8 * hh);
  f.q[1] = *(const uint4*)(p + 16 + 8 * hh);
  return f.v;
}

// Guarded A-fragment: rows >= rowmax read as zero (K/V projection padding).
__device__ __forceinline__ v16bf ld_afrag_guard(const bf16_t* base, int pitch,
                                                int row0, int rowmax,
                                                int col0, int lane) {
  const int m = lane & 15, hh = lane >> 4;
  const int r = row0 + m;
  FragBF f;
  if (r < rowmax) {
    const bf16_t* p = base + (size_t)r * pitch + col0;
    f.q[0] = *(const uint4*)(p + 8 * hh);
    f.q[1] = *(const uint4*)(p + 16 + 8 * hh);
  } else {
    f.q[0] = make_uint4(0, 0, 0, 0);
    f.q[1] = make_uint4(0, 0, 0, 0);
  }
  return f.v;
}

// B-fragment (32x16 bf16, K x N) from an N-major ("B-transposed") buffer:
// lanes 0-15 -> K {0..7, 16..23}; lanes 16-31 -> K {8..15, 24..31}.
__device__ __forceinline__ v16bf ld_bfrag(const bf16_t* base, int pitch,
                                          int nrow0, int k0, int lane) {
  const int n = lane & 15, hh = lane >> 4;
  const bf16_t* p = base + (size_t)(nrow0 + n) * pitch + k0;
  FragBF f;
  f.q[0] = *(const uint4*)(p + 8 * hh);
  f.q[1] = *(const uint4*)(p + 16 + 8 * hh);
  return f.v;
}

// B-fragment with upper-half K zeroed (used where K 80..95 is padding).
__device__ __forceinline__ v16bf ld_bfrag_zhi(const bf16_t* base, int pitch,
                                              int nrow0, int k0, int lane) {
  const int n = lane & 15, hh = lane >> 4;
  FragBF f;
  const bf16_t* p = base + (size_t)(nrow0 + n) * pitch + k0;
  f.q[0] = *(const uint4*)(p + 8 * hh);
  f.q[1] = make_uint4(0, 0, 0, 0);
  return f.v;
}

__device__ __forceinline__ v8f wmma_bf16(v16bf a, v16bf b, v8f c) {
  return __builtin_amdgcn_wmma_f32_16x16x32_bf16(
      /*neg_a=*/false, a, /*neg_b=*/false, b,
      /*c_mod=*/(short)0, c, /*reuse_a=*/false, /*reuse_b=*/false);
}

// ---------------------------------------------------------------------------
// Tensor Data Mover: 2D tile load, global -> LDS (D# per ISA 08_async_tensor).
// group0: count=1 | lds_addr | global_addr(57b) | type=2
// group1: data_size=2B, tensor_dim0/1, tile_dim0/1, dim0_stride
// ---------------------------------------------------------------------------
__device__ __forceinline__ void tdm_load_2d(unsigned lds_off, const void* gptr,
                                            unsigned dim0, unsigned dim1,
                                            unsigned stride0, unsigned tile0,
                                            unsigned tile1) {
  const unsigned long long ga = (unsigned long long)(uintptr_t)gptr;
  u32x4 g0 = {
      1u,                                                  // count=1
      lds_off,                                             // LDS byte address
      (unsigned)(ga & 0xFFFFFFFFu),                        // global_addr lo
      (unsigned)((ga >> 32) & 0x01FFFFFFu) | (2u << 30)    // addr hi | type=2
  };
  i32x8 g1 = {
      (int)(1u << 16),        // data_size = 2 bytes
      (int)(dim0 << 16),      // tensor_dim0[15:0]
      (int)(dim1 << 16),      // tensor_dim0 hi=0 | tensor_dim1[15:0]
      (int)(tile0 << 16),     // tensor_dim1 hi=0 | tile_dim0
      (int)(tile1),           // tile_dim1 | tile_dim2=0
      (int)(stride0),         // tensor_dim0_stride[31:0]
      0, 0                    // stride hi / dim1_stride (unused, 2D tile)
  };
  i32x4 gz = {0, 0, 0, 0};
#if __clang_major__ >= 23
  i32x8 gz8 = {0, 0, 0, 0, 0, 0, 0, 0};
  __builtin_amdgcn_tensor_load_to_lds(g0, g1, gz, gz, gz8, 0);
#else
  __builtin_amdgcn_tensor_load_to_lds(g0, g1, gz, gz, 0);
#endif
}

// ---------------------------------------------------------------------------
// Prep kernel 1: fp32 [K][N] -> bf16 [N][K] transpose (weights).
// ---------------------------------------------------------------------------
__global__ __launch_bounds__(256) void xattn_transpose_bf16(
    const float* __restrict__ src, bf16_t* __restrict__ dst, int K, int Nn) {
  __shared__ float tile[32][33];
  const int bk = blockIdx.x * 32, bn = blockIdx.y * 32;
  const int tx = threadIdx.x, ty = threadIdx.y;  // 32 x 8
#pragma unroll
  for (int i = 0; i < 32; i += 8)
    tile[ty + i][tx] = src[(size_t)(bk + ty + i) * Nn + bn + tx];
  __syncthreads();
#pragma unroll
  for (int i = 0; i < 32; i += 8)
    dst[(size_t)(bn + ty + i) * K + bk + tx] = f2bf(tile[tx][ty + i]);
}

// Prep kernel 2: fp32 -> bf16 elementwise (cond).
__global__ __launch_bounds__(256) void xattn_convert_bf16(
    const float* __restrict__ src, bf16_t* __restrict__ dst, int n) {
  const int i = blockIdx.x * blockDim.x + threadIdx.x;
  if (i < n) dst[i] = f2bf(src[i]);
}

// ---------------------------------------------------------------------------
// K/V projection: cond_bf [B*77, 768] @ Wk/Wv -> kpad [B,H,96,64] bf16 and
// vpadT [B,H,64,96] bf16 (V stored d-major so P@V B-fragments are contiguous).
// Rows >= 77 stored as zero. grid = B*2*6, block = 256 (wave == head).
// ---------------------------------------------------------------------------
__global__ __launch_bounds__(256) void xattn_kv_proj(
    const bf16_t* __restrict__ cond_bf, const bf16_t* __restrict__ WkT,
    const bf16_t* __restrict__ WvT, bf16_t* __restrict__ kpad,
    bf16_t* __restrict__ vpadT) {
  const int bid = blockIdx.x;
  const int b   = bid / 12;
  const int rem = bid % 12;
  const int mat = rem / 6;   // 0 = K, 1 = V
  const int rt  = rem % 6;   // 16-row tile of padded M (0..95)
  const int wid  = threadIdx.x >> 5;   // head
  const int lane = threadIdx.x & 31;
  const int m16 = lane & 15, hh = lane >> 4;
  const bf16_t* WT = (mat == 0) ? WkT : WvT;

  v8f acc[4];
#pragma unroll
  for (int nt = 0; nt < 4; ++nt)
    acc[nt] = v8f{0.f, 0.f, 0.f, 0.f, 0.f, 0.f, 0.f, 0.f};

  if (rt < 5) {  // rows 80..95 are pure padding, skip compute
    for (int ks = 0; ks < COND_D / 32; ++ks) {
      const int k0 = ks * 32;
      v16bf a = ld_afrag_guard(cond_bf, COND_D, b * SEQ_M + rt * 16,
                               b * SEQ_M + SEQ_M, k0, lane);
#pragma unroll
      for (int nt = 0; nt < 4; ++nt) {
        v16bf bf = ld_bfrag(WT, COND_D, wid * DHEAD + nt * 16, k0, lane);
        acc[nt] = wmma_bf16(a, bf, acc[nt]);
      }
    }
  }

#pragma unroll
  for (int nt = 0; nt < 4; ++nt) {
    const int d = nt * 16 + m16;
#pragma unroll
    for (int r = 0; r < 8; ++r) {
      const int m = rt * 16 + r + 8 * hh;
      const float val = (m < SEQ_M) ? acc[nt][r] : 0.0f;
      if (mat == 0) {
        kpad[((size_t)(b * NHEADS + wid) * MPAD + m) * DHEAD + d] = f2bf(val);
      } else {
        vpadT[((size_t)(b * NHEADS + wid) * DHEAD + d) * MPAD + m] = f2bf(val);
      }
    }
  }
}

// ---------------------------------------------------------------------------
// Fused main kernel (512 WGs x 256 threads, wave w == head w in phase 3).
// ---------------------------------------------------------------------------
__global__ __launch_bounds__(256, 1) void xattn_fused(
    const float* __restrict__ x, const bf16_t* __restrict__ WqT,
    const bf16_t* __restrict__ WoT, const float* __restrict__ bo,
    const bf16_t* __restrict__ kpad, const bf16_t* __restrict__ vpadT,
    float* __restrict__ out) {
  extern __shared__ char smem_raw[];
  bf16_t* QO = (bf16_t*)smem_raw;  // 128 x QO_PITCH

  const int tid  = threadIdx.x;
  const int wid  = tid >> 5;
  const int lane = tid & 31;
  const int row0 = blockIdx.x * ROWS_WG;       // global row in [0, B*N)
  const int b    = row0 >> 12;                 // / SEQ_N
  const int m16 = lane & 15, hh = lane >> 4;

  // ---- Issue TDM loads of this wave's head K/V tiles; overlaps phase 1 ----
  {
    const bf16_t* kg = kpad + (size_t)(b * NHEADS + wid) * MPAD * DHEAD;
    const bf16_t* vg = vpadT + (size_t)(b * NHEADS + wid) * DHEAD * MPAD;
    const unsigned k_lds =
        (unsigned)(uintptr_t)(void*)(smem_raw + K_OFF + wid * KLDS_HEAD);
    const unsigned v_lds =
        (unsigned)(uintptr_t)(void*)(smem_raw + V_OFF + wid * VLDS_HEAD);
    // K: tensor [96 x 64] stride 64, tile 64 x 77 rows -> LDS 77x64 packed
    tdm_load_2d(k_lds, kg, DHEAD, MPAD, DHEAD, DHEAD, KLDS_ROWS);
    // V^T: tensor [64 x 96] stride 96, tile 80 x 64 rows -> LDS 64x80 packed
    tdm_load_2d(v_lds, vg, MPAD, DHEAD, MPAD, VLDS_PITCH, DHEAD);
  }

  // ---------------- Phase 1: Q projection (barrier-free) ----------------
  // Wave w owns rows [w*16, w*16+16). All 16 A-fragments (K=512) are loaded
  // from x (f32), converted to bf16 and kept in registers (128 VGPRs).
  {
    v16bf aq[16];
#pragma unroll
    for (int ks = 0; ks < 16; ++ks) {
      const float* p =
          x + (size_t)(row0 + wid * 16 + m16) * FEAT + ks * 32 + 8 * hh;
      float4 f0 = *(const float4*)(p);
      float4 f1 = *(const float4*)(p + 4);
      float4 f2 = *(const float4*)(p + 16);
      float4 f3 = *(const float4*)(p + 20);
      FragBF fb;
      fb.q[0].x = pack2(f0.x, f0.y); fb.q[0].y = pack2(f0.z, f0.w);
      fb.q[0].z = pack2(f1.x, f1.y); fb.q[0].w = pack2(f1.z, f1.w);
      fb.q[1].x = pack2(f2.x, f2.y); fb.q[1].y = pack2(f2.z, f2.w);
      fb.q[1].z = pack2(f3.x, f3.y); fb.q[1].w = pack2(f3.z, f3.w);
      aq[ks] = fb.v;
    }
    for (int cb = 0; cb < 4; ++cb) {  // 128-column blocks of D_ATTN
      v8f acc[8];
#pragma unroll
      for (int ct = 0; ct < 8; ++ct)
        acc[ct] = v8f{0.f, 0.f, 0.f, 0.f, 0.f, 0.f, 0.f, 0.f};
#pragma unroll
      for (int ks = 0; ks < 16; ++ks) {
#pragma unroll
        for (int ct = 0; ct < 8; ++ct) {
          v16bf bfr = ld_bfrag(WqT, FEAT, cb * 128 + ct * 16, ks * 32, lane);
          acc[ct] = wmma_bf16(aq[ks], bfr, acc[ct]);
        }
      }
#pragma unroll
      for (int ct = 0; ct < 8; ++ct) {
        const int col = cb * 128 + ct * 16 + m16;
        const int rbase = wid * 16 + 8 * hh;
#pragma unroll
        for (int r = 0; r < 8; ++r)
          QO[(size_t)(rbase + r) * QO_PITCH + col] = f2bf(acc[ct][r]);
      }
    }
  }
  __syncthreads();  // all Q rows visible to all waves

  // ---------------- Phase 3: attention (wave == head) ----------------
  {
    const int h = wid;
    // Wait for this wave's own TDM transfers; no barrier needed (each wave
    // reads only the LDS region its own tensor loads filled).
    __builtin_amdgcn_s_wait_tensorcnt(0);

    const bf16_t* kLDS = (const bf16_t*)(smem_raw + K_OFF + h * KLDS_HEAD);
    const bf16_t* vLDS = (const bf16_t*)(smem_raw + V_OFF + h * VLDS_HEAD);

    v16bf kf[2][5], vf[3][4];
#pragma unroll
    for (int kk = 0; kk < 2; ++kk)
#pragma unroll
      for (int mt = 0; mt < 5; ++mt)
        kf[kk][mt] = ld_bfrag(kLDS, DHEAD, mt * 16, kk * 32, lane);
#pragma unroll
    for (int k3 = 0; k3 < 3; ++k3)
#pragma unroll
      for (int nt = 0; nt < 4; ++nt)
        vf[k3][nt] = (k3 == 2)
                         ? ld_bfrag_zhi(vLDS, VLDS_PITCH, nt * 16, 64, lane)
                         : ld_bfrag(vLDS, VLDS_PITCH, nt * 16, k3 * 32, lane);

    bf16_t* P = (bf16_t*)(smem_raw + P_OFF) + (size_t)wid * 16 * P_PITCH;
    {  // zero P columns 80..95 once (K-dim padding for P@V)
      const int prow = m16, pcol = 80 + hh * 8;
#pragma unroll
      for (int r = 0; r < 8; ++r) P[prow * P_PITCH + pcol + r] = f2bf(0.0f);
    }

    const float scale = 0.125f;  // 1/sqrt(64)
    for (int rt = 0; rt < 8; ++rt) {
      // S = q @ k^T (16 rows x 80, cols >= 77 masked below)
      v8f s[5];
#pragma unroll
      for (int mt = 0; mt < 5; ++mt)
        s[mt] = v8f{0.f, 0.f, 0.f, 0.f, 0.f, 0.f, 0.f, 0.f};
#pragma unroll
      for (int kk = 0; kk < 2; ++kk) {
        v16bf a = ld_afrag(QO, QO_PITCH, rt * 16, h * DHEAD + kk * 32, lane);
#pragma unroll
        for (int mt = 0; mt < 5; ++mt) s[mt] = wmma_bf16(a, kf[kk][mt], s[mt]);
      }
      // softmax over M; row r+8*hh lives in the 16-lane half `hh`
#pragma unroll
      for (int r = 0; r < 8; ++r) {
        float vals[5];
        float mx = -3.0e38f;
#pragma unroll
        for (int mt = 0; mt < 5; ++mt) {
          const int col = mt * 16 + m16;
          vals[mt] = (col < SEQ_M) ? s[mt][r] * scale : -3.0e38f;
          mx = fmaxf(mx, vals[mt]);
        }
#pragma unroll
        for (int off = 8; off >= 1; off >>= 1)
          mx = fmaxf(mx, __shfl_xor(mx, off, 32));
        float sum = 0.0f;
#pragma unroll
        for (int mt = 0; mt < 5; ++mt) {
          const int col = mt * 16 + m16;
          const float e = (col < SEQ_M) ? __expf(vals[mt] - mx) : 0.0f;
          vals[mt] = e;
          sum += e;
        }
#pragma unroll
        for (int off = 8; off >= 1; off >>= 1) sum += __shfl_xor(sum, off, 32);
        const float rs = 1.0f / sum;
        const int prow = r + 8 * hh;
#pragma unroll
        for (int mt = 0; mt < 5; ++mt)
          P[prow * P_PITCH + mt * 16 + m16] = f2bf(vals[mt] * rs);
      }
      // O = P @ V (16 x 64)
      v8f o[4];
#pragma unroll
      for (int nt = 0; nt < 4; ++nt)
        o[nt] = v8f{0.f, 0.f, 0.f, 0.f, 0.f, 0.f, 0.f, 0.f};
#pragma unroll
      for (int k3 = 0; k3 < 3; ++k3) {
        v16bf a = ld_afrag(P, P_PITCH, 0, k3 * 32, lane);
#pragma unroll
        for (int nt = 0; nt < 4; ++nt) o[nt] = wmma_bf16(a, vf[k3][nt], o[nt]);
      }
      // overwrite Q with O in place (only wave h touches cols h*64..h*64+63)
#pragma unroll
      for (int nt = 0; nt < 4; ++nt) {
        const int col = h * DHEAD + nt * 16 + m16;
        const int rbase = rt * 16 + 8 * hh;
#pragma unroll
        for (int r = 0; r < 8; ++r)
          QO[(size_t)(rbase + r) * QO_PITCH + col] = f2bf(o[nt][r]);
      }
    }
  }
  __syncthreads();  // all O columns visible to all waves

  // ---------------- Phase 4: out = O @ Wo + bo ----------------
  {
    const int wave_r = wid >> 2;  // 0..1 (64-row band)
    const int wave_c = wid & 3;   // 0..3 (32-col band)
    for (int cb = 0; cb < 4; ++cb) {
      v8f acc[4][2];
#pragma unroll
      for (int rt = 0; rt < 4; ++rt)
#pragma unroll
        for (int ct = 0; ct < 2; ++ct)
          acc[rt][ct] = v8f{0.f, 0.f, 0.f, 0.f, 0.f, 0.f, 0.f, 0.f};

      for (int ks = 0; ks < FEAT / 32; ++ks) {
        const int k0 = ks * 32;
        v16bf bfr[2];
#pragma unroll
        for (int ct = 0; ct < 2; ++ct)
          bfr[ct] =
              ld_bfrag(WoT, FEAT, cb * 128 + wave_c * 32 + ct * 16, k0, lane);
#pragma unroll
        for (int rt = 0; rt < 4; ++rt) {
          v16bf a = ld_afrag(QO, QO_PITCH, wave_r * 64 + rt * 16, k0, lane);
#pragma unroll
          for (int ct = 0; ct < 2; ++ct)
            acc[rt][ct] = wmma_bf16(a, bfr[ct], acc[rt][ct]);
        }
      }
#pragma unroll
      for (int rt = 0; rt < 4; ++rt)
#pragma unroll
        for (int ct = 0; ct < 2; ++ct) {
          const int col = cb * 128 + wave_c * 32 + ct * 16 + m16;
          const float bias = bo[col];
          const int rbase = wave_r * 64 + rt * 16 + 8 * hh;
#pragma unroll
          for (int r = 0; r < 8; ++r)
            out[(size_t)(row0 + rbase + r) * FEAT + col] =
                acc[rt][ct][r] + bias;
        }
    }
  }
}

// ---------------------------------------------------------------------------
// Host launcher. Inputs: x, cond, Wq, Wk, Wv, Wo, bo (all f32). Output f32.
// ---------------------------------------------------------------------------
extern "C" void kernel_launch(void* const* d_in, const int* in_sizes, int n_in,
                              void* d_out, int out_size, void* d_ws,
                              size_t ws_size, hipStream_t stream) {
  (void)in_sizes; (void)n_in; (void)out_size; (void)ws_size;
  const float* x    = (const float*)d_in[0];
  const float* cond = (const float*)d_in[1];
  const float* Wq   = (const float*)d_in[2];
  const float* Wk   = (const float*)d_in[3];
  const float* Wv   = (const float*)d_in[4];
  const float* Wo   = (const float*)d_in[5];
  const float* bo   = (const float*)d_in[6];
  float* out = (float*)d_out;

  // Workspace layout (bf16 regions, ~7.7 MB total)
  char* ws = (char*)d_ws;
  bf16_t* WqT     = (bf16_t*)(ws);            // 512*512
  bf16_t* WkT     = (bf16_t*)(ws + 524288);   // 512*768
  bf16_t* WvT     = (bf16_t*)(ws + 1310720);  // 512*768
  bf16_t* WoT     = (bf16_t*)(ws + 2097152);  // 512*512
  bf16_t* cond_bf = (bf16_t*)(ws + 2621440);  // 16*77*768
  bf16_t* kpad    = (bf16_t*)(ws + 4513792);  // 16*8*96*64
  bf16_t* vpadT   = (bf16_t*)(ws + 6086656);  // 16*8*64*96

  // Prep: weight transpose+convert, cond convert.
  xattn_transpose_bf16<<<dim3(FEAT / 32, 512 / 32), dim3(32, 8), 0, stream>>>(
      Wq, WqT, FEAT, 512);
  xattn_transpose_bf16<<<dim3(COND_D / 32, 512 / 32), dim3(32, 8), 0, stream>>>(
      Wk, WkT, COND_D, 512);
  xattn_transpose_bf16<<<dim3(COND_D / 32, 512 / 32), dim3(32, 8), 0, stream>>>(
      Wv, WvT, COND_D, 512);
  xattn_transpose_bf16<<<dim3(FEAT / 32, 512 / 32), dim3(32, 8), 0, stream>>>(
      Wo, WoT, FEAT, 512);
  {
    const int n = BATCH * SEQ_M * COND_D;
    xattn_convert_bf16<<<(n + 255) / 256, 256, 0, stream>>>(cond, cond_bf, n);
  }

  // K/V projections into padded, attention-friendly layouts.
  xattn_kv_proj<<<BATCH * 2 * 6, 256, 0, stream>>>(cond_bf, WkT, WvT, kpad,
                                                   vpadT);

  // Fused Q-proj -> attention -> O-proj (TDM prefetch of K/V inside).
  (void)hipFuncSetAttribute((const void*)xattn_fused,
                            hipFuncAttributeMaxDynamicSharedMemorySize,
                            SMEM_BYTES);
  xattn_fused<<<BATCH * SEQ_N / ROWS_WG, 256, SMEM_BYTES, stream>>>(
      x, WqT, WoT, bo, kpad, vpadT, out);
}
